// MoE_12077448037043
// MI455X (gfx1250) — compile-verified, matrix operand order checked by reading, not run on hardware
//
#include <hip/hip_runtime.h>
#include <hip/hip_bf16.h>

// MoE top-2, T=2048 D=2048 F=4096 E=8.
// Sparse grouped-GEMM formulation; fp32 weights converted to bf16 while
// staging into LDS, math via V_WMMA_F32_16X16X32_BF16 (wave32).
// Down-proj A operand (bf16 H) staged with GLOBAL_LOAD_ASYNC_TO_LDS_B128.

constexpr int Tt = 2048;
constexpr int Dd = 2048;
constexpr int Ff = 4096;
constexpr int Ee = 8;

typedef __bf16 bf16_t;
typedef __bf16 v16bf __attribute__((ext_vector_type(16)));
typedef __bf16 v8bf  __attribute__((ext_vector_type(8)));
typedef __bf16 v4bf  __attribute__((ext_vector_type(4)));
typedef float  v8f   __attribute__((ext_vector_type(8)));
typedef float  v4f   __attribute__((ext_vector_type(4)));

__device__ __forceinline__ bf16_t f2bf(float f) {
    unsigned u = __builtin_bit_cast(unsigned, f);
    u += 0x7FFFu + ((u >> 16) & 1u);              // round-to-nearest-even
    unsigned short s = (unsigned short)(u >> 16);
    return __builtin_bit_cast(bf16_t, s);
}

// pack two floats as bf16 pair {lo, hi} into one dword (RNE)
__device__ __forceinline__ unsigned pk2bf(float lo, float hi) {
    unsigned ua = __builtin_bit_cast(unsigned, lo);
    ua += 0x7FFFu + ((ua >> 16) & 1u);
    unsigned ub = __builtin_bit_cast(unsigned, hi);
    ub += 0x7FFFu + ((ub >> 16) & 1u);
    return (ua >> 16) | (ub & 0xFFFF0000u);
}

__device__ __forceinline__ v16bf frag_from_lds(const bf16_t* p) {
    // p points at this lane's K-chunk start (16B aligned).
    // VGPRs 0..3 <- K k0..k0+7 ; VGPRs 4..7 <- K k0+16..k0+23
    v8bf c0 = *(const v8bf*)p;
    v8bf c1 = *(const v8bf*)(p + 16);
    return __builtin_shufflevector(c0, c1, 0,1,2,3,4,5,6,7,8,9,10,11,12,13,14,15);
}

// ---- CDNA5 async global->LDS copy (ASYNCcnt tracked) --------------------
typedef __attribute__((address_space(3))) char* lds_as3_t;

__device__ __forceinline__ unsigned lds_off_of(void* p) {
    return (unsigned)(size_t)(lds_as3_t)p;
}

__device__ __forceinline__ void async_ld_b128(void* lds_dst, const void* gsrc) {
    unsigned off = lds_off_of(lds_dst);
    asm volatile("global_load_async_to_lds_b128 %0, %1, off"
                 :: "v"(off), "v"(gsrc)
                 : "memory");
}

__device__ __forceinline__ void wait_async0() {
#if __has_builtin(__builtin_amdgcn_s_wait_asynccnt)
    __builtin_amdgcn_s_wait_asynccnt(0);
#else
    asm volatile("s_wait_asynccnt 0x0" ::: "memory");
#endif
}

// ---------------------------------------------------------------- utility
__global__ void zero_cnt_kernel(int* cnt) {
    if (threadIdx.x < Ee) cnt[threadIdx.x] = 0;
}

__global__ void offsets_kernel(const int* __restrict__ cnt, int* __restrict__ hb) {
    if (threadIdx.x == 0) {
        int b = 0;
        for (int e = 0; e < Ee; ++e) { hb[e] = b; b += cnt[e]; }
    }
}

// ---------------------------------------------------------------- router
__global__ __launch_bounds__(256)
void router_kernel(const float* __restrict__ x, const float* __restrict__ rw,
                   int* __restrict__ cnt, int* __restrict__ tok_of,
                   float* __restrict__ w_of, int* __restrict__ ex_tk,
                   int* __restrict__ pos_tk) {
    const int wave = threadIdx.x >> 5;
    const int lane = threadIdx.x & 31;
    const int t = blockIdx.x * 8 + wave;

    float acc[Ee];
#pragma unroll
    for (int e = 0; e < Ee; ++e) acc[e] = 0.f;

    const float* xr = x + (size_t)t * Dd;
    for (int d = lane; d < Dd; d += 32) {
        float xv = xr[d];
        const float* r = rw + (size_t)d * Ee;
        v4f r0 = *(const v4f*)r;
        v4f r1 = *(const v4f*)(r + 4);
        acc[0] += xv * r0.x; acc[1] += xv * r0.y;
        acc[2] += xv * r0.z; acc[3] += xv * r0.w;
        acc[4] += xv * r1.x; acc[5] += xv * r1.y;
        acc[6] += xv * r1.z; acc[7] += xv * r1.w;
    }
#pragma unroll
    for (int e = 0; e < Ee; ++e) {
#pragma unroll
        for (int off = 16; off >= 1; off >>= 1)
            acc[e] += __shfl_xor(acc[e], off, 32);
    }

    if (lane == 0) {
        int e0 = 0; float l0 = acc[0];
#pragma unroll
        for (int e = 1; e < Ee; ++e) if (acc[e] > l0) { l0 = acc[e]; e0 = e; }
        int e1 = -1; float l1 = -3.4e38f;
#pragma unroll
        for (int e = 0; e < Ee; ++e)
            if (e != e0 && acc[e] > l1) { l1 = acc[e]; e1 = e; }
        float w0 = 1.f / (1.f + __expf(l1 - l0));
        float w1 = 1.f - w0;
        int p0 = atomicAdd(&cnt[e0], 1);
        int p1 = atomicAdd(&cnt[e1], 1);
        tok_of[e0 * Tt + p0] = t;  w_of[e0 * Tt + p0] = w0;
        tok_of[e1 * Tt + p1] = t;  w_of[e1 * Tt + p1] = w1;
        ex_tk[2 * t + 0] = e0;  pos_tk[2 * t + 0] = p0;
        ex_tk[2 * t + 1] = e1;  pos_tk[2 * t + 1] = p1;
    }
}

// ---------------------------------------------------------------- gate/up
// Tile 128(M) x 64(N), K=32 chunks.  8 waves = 4(M) x 2(N); each wave owns
// 2x2 WMMA tiles for BOTH gate and up (A reuse).  H = silu(g)*u -> bf16.
__global__ __launch_bounds__(256)
void gateup_kernel(const float* __restrict__ x,
                   const float* __restrict__ wg,
                   const float* __restrict__ wu,
                   const int* __restrict__ cnt,
                   const int* __restrict__ hb,
                   const int* __restrict__ tok_of,
                   bf16_t* __restrict__ H) {
    const int e = blockIdx.z;
    const int count = cnt[e];
    const int m0 = blockIdx.y * 128;
    if (m0 >= count) return;
    const int n0 = blockIdx.x * 64;
    const int hbase = hb[e];
    const size_t wbase = (size_t)e * Dd * Ff;
    const float* wgp = wg + wbase;
    const float* wup = wu + wbase;

    __shared__ bf16_t ldsA[128 * 40];   // row-major [row][k], stride 40
    __shared__ bf16_t ldsBg[64 * 40];   // col-major [col][k], stride 40
    __shared__ bf16_t ldsBu[64 * 40];

    const int tid = threadIdx.x;
    const int lane = tid & 31;
    const int waveId = tid >> 5;
    const int wm = waveId & 3;   // M quadrant (32 rows)
    const int wn = waveId >> 2;  // N half (32 cols)
    const int l15 = lane & 15;
    const int hi = lane >> 4;    // K-half select

    // gathered rows this thread stages: clamped token + validity mask
    // (unconditional loads; garbage rows zeroed via multiply -> no EXEC churn)
    int   tokc[4];
    float msk[4];
#pragma unroll
    for (int j = 0; j < 4; ++j) {
        int r = (tid >> 3) + 32 * j;
        int pos = m0 + r;
        tokc[j] = tok_of[e * Tt + pos] & (Tt - 1);   // always in-bounds
        msk[j] = (pos < count) ? 1.f : 0.f;
    }
    const int kq = tid & 7;      // float4 index within 32 K values

    // B staging decomposition: thread -> (k pair, f4)
    const int bk = (tid >> 4) * 2;   // 0,2,..,30
    const int bf4 = tid & 15;        // 0..15

    v8f accg[2][2] = {};
    v8f accu[2][2] = {};

    for (int kb = 0; kb < Dd; kb += 32) {
        // ---- stage A (gathered x rows, f32 -> bf16), unconditional
#pragma unroll
        for (int j = 0; j < 4; ++j) {
            int r = (tid >> 3) + 32 * j;
            v4f v = *(const v4f*)(x + (size_t)tokc[j] * Dd + kb + kq * 4);
            float m = msk[j];
            v4bf pk;
            pk.x = f2bf(v.x * m); pk.y = f2bf(v.y * m);
            pk.z = f2bf(v.z * m); pk.w = f2bf(v.w * m);
            *(v4bf*)(&ldsA[r * 40 + kq * 4]) = pk;
        }
        // ---- stage B gate/up: K-pair packed dword stores into col-major LDS
        {
            size_t g0 = (size_t)(kb + bk) * Ff + n0 + bf4 * 4;
            v4f g_lo = *(const v4f*)(wgp + g0);
            v4f g_hi = *(const v4f*)(wgp + g0 + Ff);
            v4f u_lo = *(const v4f*)(wup + g0);
            v4f u_hi = *(const v4f*)(wup + g0 + Ff);
            int cb = bf4 * 4;
            *(unsigned*)&ldsBg[(cb + 0) * 40 + bk] = pk2bf(g_lo.x, g_hi.x);
            *(unsigned*)&ldsBg[(cb + 1) * 40 + bk] = pk2bf(g_lo.y, g_hi.y);
            *(unsigned*)&ldsBg[(cb + 2) * 40 + bk] = pk2bf(g_lo.z, g_hi.z);
            *(unsigned*)&ldsBg[(cb + 3) * 40 + bk] = pk2bf(g_lo.w, g_hi.w);
            *(unsigned*)&ldsBu[(cb + 0) * 40 + bk] = pk2bf(u_lo.x, u_hi.x);
            *(unsigned*)&ldsBu[(cb + 1) * 40 + bk] = pk2bf(u_lo.y, u_hi.y);
            *(unsigned*)&ldsBu[(cb + 2) * 40 + bk] = pk2bf(u_lo.z, u_hi.z);
            *(unsigned*)&ldsBu[(cb + 3) * 40 + bk] = pk2bf(u_lo.w, u_hi.w);
        }
        // ---- prefetch next weight K-chunk (global_prefetch_b8)
        if (kb + 32 < Dd) {
            size_t np = (size_t)(kb + 32 + (tid >> 3)) * Ff + n0 + (tid & 7) * 8;
            __builtin_prefetch(wgp + np, 0, 1);
            __builtin_prefetch(wup + np, 0, 1);
        }
        __syncthreads();

        // ---- WMMA
#pragma unroll
        for (int mt = 0; mt < 2; ++mt) {
            const bf16_t* ap = &ldsA[(wm * 32 + mt * 16 + l15) * 40 + hi * 8];
            v16bf a = frag_from_lds(ap);
#pragma unroll
            for (int nt = 0; nt < 2; ++nt) {
                const int cbase = (wn * 32 + nt * 16 + l15) * 40 + hi * 8;
                v16bf bgf = frag_from_lds(&ldsBg[cbase]);
                accg[mt][nt] = __builtin_amdgcn_wmma_f32_16x16x32_bf16(
                    false, a, false, bgf, (short)0, accg[mt][nt], false, false);
                v16bf buf = frag_from_lds(&ldsBu[cbase]);
                accu[mt][nt] = __builtin_amdgcn_wmma_f32_16x16x32_bf16(
                    false, a, false, buf, (short)0, accu[mt][nt], false, false);
            }
        }
        __syncthreads();
    }

    // ---- epilogue: h = silu(g) * u  -> bf16 H[slot][col]
#pragma unroll
    for (int mt = 0; mt < 2; ++mt) {
#pragma unroll
        for (int nt = 0; nt < 2; ++nt) {
            int col = n0 + wn * 32 + nt * 16 + l15;
#pragma unroll
            for (int i = 0; i < 8; ++i) {
                int pos = m0 + wm * 32 + mt * 16 + hi * 8 + i;
                if (pos < count) {
                    float g = accg[mt][nt][i];
                    float u = accu[mt][nt][i];
                    float h = (g / (1.f + __expf(-g))) * u;
                    H[(size_t)(hbase + pos) * Ff + col] = f2bf(h);
                }
            }
        }
    }
}

// ---------------------------------------------------------------- down
// contrib[slot] = router_weight * (H[slot] @ w_down[e]).  A operand (bf16 H)
// staged with async global->LDS copies on full tiles.
__global__ __launch_bounds__(256)
void down_kernel(const bf16_t* __restrict__ H,
                 const float* __restrict__ wd,
                 const int* __restrict__ cnt,
                 const int* __restrict__ hb,
                 const float* __restrict__ w_of,
                 float* __restrict__ contrib) {
    const int e = blockIdx.z;
    const int count = cnt[e];
    const int m0 = blockIdx.y * 128;
    if (m0 >= count) return;
    const int n0 = blockIdx.x * 64;
    const int hbase = hb[e];
    const float* wdp = wd + (size_t)e * Ff * Dd;
    const bool fullTile = (m0 + 128 <= count);

    __shared__ bf16_t ldsA[128 * 40];
    __shared__ bf16_t ldsB[64 * 40];

    const int tid = threadIdx.x;
    const int lane = tid & 31;
    const int waveId = tid >> 5;
    const int wm = waveId & 3;
    const int wn = waveId >> 2;
    const int l15 = lane & 15;
    const int hi = lane >> 4;

    const int bk = (tid >> 4) * 2;
    const int bf4 = tid & 15;

    v8f acc[2][2] = {};

    for (int kb = 0; kb < Ff; kb += 32) {
        // ---- stage A from H (already bf16)
        if (fullTile) {
            // CDNA5 async copy: 16B per lane, ASYNCcnt tracked
#pragma unroll
            for (int j = 0; j < 2; ++j) {
                int c = tid + 256 * j;        // 0..511
                int r = c >> 2;               // 0..127
                int kc = c & 3;               // 8-half chunk
                const bf16_t* src =
                    H + (size_t)(hbase + m0 + r) * Ff + kb + kc * 8;
                async_ld_b128(&ldsA[r * 40 + kc * 8], src);
            }
        } else {
#pragma unroll
            for (int j = 0; j < 2; ++j) {
                int c = tid + 256 * j;
                int r = c >> 2;
                int kc = c & 3;
                int pos = m0 + r;
                v8bf v = {};
                if (pos < count)
                    v = *(const v8bf*)(H + (size_t)(hbase + pos) * Ff + kb + kc * 8);
                *(v8bf*)(&ldsA[r * 40 + kc * 8]) = v;
            }
        }
        // ---- stage B from w_down: K-pair packed dword stores
        {
            size_t g0 = (size_t)(kb + bk) * Dd + n0 + bf4 * 4;
            v4f v_lo = *(const v4f*)(wdp + g0);
            v4f v_hi = *(const v4f*)(wdp + g0 + Dd);
            int cb = bf4 * 4;
            *(unsigned*)&ldsB[(cb + 0) * 40 + bk] = pk2bf(v_lo.x, v_hi.x);
            *(unsigned*)&ldsB[(cb + 1) * 40 + bk] = pk2bf(v_lo.y, v_hi.y);
            *(unsigned*)&ldsB[(cb + 2) * 40 + bk] = pk2bf(v_lo.z, v_hi.z);
            *(unsigned*)&ldsB[(cb + 3) * 40 + bk] = pk2bf(v_lo.w, v_hi.w);
        }
        if (kb + 32 < Ff) {
            size_t np = (size_t)(kb + 32 + (tid >> 3)) * Dd + n0 + (tid & 7) * 8;
            __builtin_prefetch(wdp + np, 0, 1);
        }
        if (fullTile) wait_async0();
        __syncthreads();

#pragma unroll
        for (int mt = 0; mt < 2; ++mt) {
            v16bf a = frag_from_lds(&ldsA[(wm * 32 + mt * 16 + l15) * 40 + hi * 8]);
#pragma unroll
            for (int nt = 0; nt < 2; ++nt) {
                v16bf b = frag_from_lds(&ldsB[(wn * 32 + nt * 16 + l15) * 40 + hi * 8]);
                acc[mt][nt] = __builtin_amdgcn_wmma_f32_16x16x32_bf16(
                    false, a, false, b, (short)0, acc[mt][nt], false, false);
            }
        }
        __syncthreads();
    }

    // ---- epilogue: scale by router weight, store per-slot contribution
#pragma unroll
    for (int mt = 0; mt < 2; ++mt) {
#pragma unroll
        for (int nt = 0; nt < 2; ++nt) {
            int col = n0 + wn * 32 + nt * 16 + l15;
#pragma unroll
            for (int i = 0; i < 8; ++i) {
                int pos = m0 + wm * 32 + mt * 16 + hi * 8 + i;
                if (pos < count) {
                    float w = w_of[e * Tt + pos];
                    contrib[(size_t)(hbase + pos) * Dd + col] = acc[mt][nt][i] * w;
                }
            }
        }
    }
}

// ---------------------------------------------------------------- combine
__global__ __launch_bounds__(256)
void combine_kernel(const float* __restrict__ contrib,
                    const int* __restrict__ ex_tk,
                    const int* __restrict__ pos_tk,
                    const int* __restrict__ hb,
                    float* __restrict__ out) {
    int idx = blockIdx.x * 256 + threadIdx.x;   // over T * (D/4)
    int t = idx >> 9;                            // D/4 = 512
    int d4 = idx & 511;
    int s0 = hb[ex_tk[2 * t + 0]] + pos_tk[2 * t + 0];
    int s1 = hb[ex_tk[2 * t + 1]] + pos_tk[2 * t + 1];
    v4f a = *(const v4f*)(contrib + (size_t)s0 * Dd + d4 * 4);
    v4f b = *(const v4f*)(contrib + (size_t)s1 * Dd + d4 * 4);
    v4f c = a + b;
    *(v4f*)(out + (size_t)t * Dd + d4 * 4) = c;
}

// ---------------------------------------------------------------- launch
extern "C" void kernel_launch(void* const* d_in, const int* in_sizes, int n_in,
                              void* d_out, int out_size, void* d_ws, size_t ws_size,
                              hipStream_t stream) {
    const float* x  = (const float*)d_in[0];
    const float* rw = (const float*)d_in[1];
    const float* wg = (const float*)d_in[2];
    const float* wu = (const float*)d_in[3];
    const float* wd = (const float*)d_in[4];
    float* out = (float*)d_out;

    char* p = (char*)d_ws;
    auto carve = [&](size_t bytes) {
        char* r = p;
        p += (bytes + 255) & ~(size_t)255;
        return r;
    };
    int*    cnt     = (int*)carve(Ee * sizeof(int));
    int*    hb      = (int*)carve(Ee * sizeof(int));
    int*    tok_of  = (int*)carve((size_t)Ee * Tt * sizeof(int));
    float*  w_of    = (float*)carve((size_t)Ee * Tt * sizeof(float));
    int*    ex_tk   = (int*)carve((size_t)2 * Tt * sizeof(int));
    int*    pos_tk  = (int*)carve((size_t)2 * Tt * sizeof(int));
    bf16_t* H       = (bf16_t*)carve((size_t)2 * Tt * Ff * sizeof(bf16_t));
    float*  contrib = (float*)carve((size_t)2 * Tt * Dd * sizeof(float));

    zero_cnt_kernel<<<1, 32, 0, stream>>>(cnt);
    router_kernel<<<Tt / 8, 256, 0, stream>>>(x, rw, cnt, tok_of, w_of, ex_tk, pos_tk);
    offsets_kernel<<<1, 32, 0, stream>>>(cnt, hb);
    gateup_kernel<<<dim3(Ff / 64, Tt / 128, Ee), 256, 0, stream>>>(
        x, wg, wu, cnt, hb, tok_of, H);
    down_kernel<<<dim3(Dd / 64, Tt / 128, Ee), 256, 0, stream>>>(
        H, wd, cnt, hb, w_of, contrib);
    combine_kernel<<<(Tt * (Dd / 4)) / 256, 256, 0, stream>>>(
        contrib, ex_tk, pos_tk, hb, out);
}
